// MultiHeadAttention_16449724745205
// MI455X (gfx1250) — matrix operand order repaired and checked by reading, not compiled, not run on hardware
//
#include <hip/hip_runtime.h>
#include <hip/hip_fp16.h>

typedef __attribute__((ext_vector_type(16))) _Float16 v16h;
typedef __attribute__((ext_vector_type(8)))  float    v8f;

#define N_HEAD 16
#define DHEAD  64
#define NSTATE 1024
#define SEQ    2048
#define ROW3   (3 * NSTATE)
#define NEG_BIG (-1e9f)
// 1/sqrt(64) * log2(e): softmax done in exp2 domain
#define QSCALE 0.18033688011112042f

// LDS row strides (f16 elements) padded to dodge bank conflicts
#define KSTRIDE 72   // 64 + 8
#define VSTRIDE 40   // 32 + 8

// ---- fragment loader: 16-bit A-matrix 16x32, contiguous k order (k = 16*hi + s, not the ISA table order) ----
// lane m (0-15): V0..7 = K 0..15 pairs (contiguous)
// lane m+16   : V0..7 = K 16..31
__device__ __forceinline__ v16h load_a16(const _Float16* row, int hi) {
    const unsigned int* r32 = (const unsigned int*)row;   // pairs of f16
    union { unsigned int u[8]; v16h v; } a;
    const int p = hi * 8;                                 // pair offset 0 or 8
#pragma unroll
    for (int i = 0; i < 4; ++i) {
        a.u[i]     = r32[p + i];        // K = hi*16 + 2i, +1   (-> one b128)
        a.u[4 + i] = r32[p + 4 + i];    // K = hi*16 + 8 + 2i  (-> one b128)
    }
    return a.v;
}

__device__ __forceinline__ v8f wmma_f16(v16h a, v16h b, v8f c) {
    return __builtin_amdgcn_wmma_f32_16x16x32_f16(false, a, false, b,
                                                  (short)0, c, false, false);
}

__global__ void __launch_bounds__(128)
flash_attn_causal_f16wmma(const float* __restrict__ x, float* __restrict__ out) {
    __shared__ _Float16 Klds[32][KSTRIDE];   // [key][d]   row-major
    __shared__ _Float16 Vlds[64][VSTRIDE];   // [d][key]   transposed

    const int tid  = threadIdx.x;
    const int wave = tid >> 5;
    const int lane = tid & 31;
    const int ln   = lane & 15;          // query / d-row / column index
    const int hi   = lane >> 4;          // lane half

    const int qblk = blockIdx.x;
    const int h    = blockIdx.y;
    const int b    = blockIdx.z;
    const int q0   = qblk * 64;
    const int qw   = q0 + wave * 16;     // wave's first query
    const int qlane = qw + ln;           // this lane's query row
    const int jdiag = (qw + 15) >> 5;    // tile index containing this wave's diagonal

    // ---- load Q strip as two B-fragments (32x16 f16), scale+log2e folded ----
    const float* xq = x + ((size_t)b * SEQ + qlane) * ROW3 + h * DHEAD;
    v16h bq[2];
#pragma unroll
    for (int c = 0; c < 2; ++c) {
#pragma unroll
        for (int i = 0; i < 8; ++i) {
            const int d = c * 32 + hi * 16 + 2 * i;
            const float2 f = *(const float2*)(xq + d);
            bq[c][2 * i]     = (_Float16)(f.x * QSCALE);
            bq[c][2 * i + 1] = (_Float16)(f.y * QSCALE);
        }
    }

    // ---- running softmax state (base-2 domain) + O^T accumulators ----
    float m_run = -1e30f;
    float l_run = 0.0f;
    v8f o[4];
#pragma unroll
    for (int dc = 0; dc < 4; ++dc) o[dc] = (v8f){0,0,0,0,0,0,0,0};

    const int ntiles = (q0 + 64) >> 5;   // causal: keys 0 .. q0+63

    // cooperative-load indexing
    const int krow  = tid >> 2;          // K: 4 threads per key row, 16 d each
    const int kdb   = (tid & 3) * 16;
    const int vkey  = (tid & 15) * 2;    // V: 2 keys x 8 d per thread
    const int vdb   = (tid >> 4) * 8;

    const float* xk = x + ((size_t)b * SEQ + krow) * ROW3 + NSTATE + h * DHEAD + kdb;
    const float* xv = x + ((size_t)b * SEQ + vkey) * ROW3 + 2 * NSTATE + h * DHEAD + vdb;

    for (int jt = 0; jt < ntiles; ++jt, xk += 32 * ROW3, xv += 32 * ROW3) {
        const int kb = jt * 32;

        __syncthreads();   // previous iteration's compute done before overwrite

        // ---- stage K (row-major f16) into LDS: 4x float4 -> 4x ds_store_b64 ----
#pragma unroll
        for (int j = 0; j < 4; ++j) {
            const float4 fk = *(const float4*)(xk + 4 * j);
            union { _Float16 hh[4]; uint2 u; } pk;
            pk.hh[0] = (_Float16)fk.x; pk.hh[1] = (_Float16)fk.y;
            pk.hh[2] = (_Float16)fk.z; pk.hh[3] = (_Float16)fk.w;
            *(uint2*)&Klds[krow][kdb + 4 * j] = pk.u;
        }
        // ---- stage V^T: 2 keys x 8 d, packed pair stores (8x ds_store_b32) ----
        {
            float va[8], vb[8];
#pragma unroll
            for (int j = 0; j < 2; ++j) {
                const float4 f0 = *(const float4*)(xv + 4 * j);
                const float4 f1 = *(const float4*)(xv + ROW3 + 4 * j);
                va[4*j+0] = f0.x; va[4*j+1] = f0.y; va[4*j+2] = f0.z; va[4*j+3] = f0.w;
                vb[4*j+0] = f1.x; vb[4*j+1] = f1.y; vb[4*j+2] = f1.z; vb[4*j+3] = f1.w;
            }
#pragma unroll
            for (int j = 0; j < 8; ++j) {
                union { _Float16 hh[2]; unsigned int u; } pv;
                pv.hh[0] = (_Float16)va[j];   // key vkey
                pv.hh[1] = (_Float16)vb[j];   // key vkey+1
                *(unsigned int*)&Vlds[vdb + j][vkey] = pv.u;
            }
        }

        __syncthreads();

        if (kb <= qw + 15) {    // wave-uniform: EXEC stays all-ones for WMMA
            // ---- S^T subtiles (scores already in base-2 domain) ----
            v8f st0 = (v8f){0,0,0,0,0,0,0,0};
            v8f st1 = (v8f){0,0,0,0,0,0,0,0};
#pragma unroll
            for (int c = 0; c < 2; ++c) {
                st0 = wmma_f16(load_a16(&Klds[ln][c * 32], hi),      bq[c], st0);
                st1 = wmma_f16(load_a16(&Klds[16 + ln][c * 32], hi), bq[c], st1);
            }

            // ---- causal mask: only the single diagonal tile needs it ----
            if (jt == jdiag) {
                const int qrel = qlane - kb - hi * 8;   // key_local = r (+16)
#pragma unroll
                for (int r = 0; r < 8; ++r) {
                    if (r > qrel)      st0[r] = NEG_BIG;
                    if (r + 16 > qrel) st1[r] = NEG_BIG;
                }
            }

            // ---- online softmax (per query = per lane column) ----
            float tmax = NEG_BIG;
#pragma unroll
            for (int r = 0; r < 8; ++r) {
                tmax = fmaxf(tmax, st0[r]);
                tmax = fmaxf(tmax, st1[r]);
            }
            tmax = fmaxf(tmax, __shfl_xor(tmax, 16, 32));

            float m_new = m_run;
            if (__any(tmax > m_run)) {          // rare after warm-up
                m_new = fmaxf(m_run, tmax);
                const float alpha = exp2f(m_run - m_new);
                l_run *= alpha;
#pragma unroll
                for (int dc = 0; dc < 4; ++dc)
#pragma unroll
                    for (int r = 0; r < 8; ++r) o[dc][r] *= alpha;
                m_run = m_new;
            }

            float lsum = 0.0f;
#pragma unroll
            for (int r = 0; r < 8; ++r) {
                st0[r] = exp2f(st0[r] - m_new);
                st1[r] = exp2f(st1[r] - m_new);
                lsum += st0[r] + st1[r];
            }
            lsum += __shfl_xor(lsum, 16, 32);
            l_run += lsum;

            // ---- assemble P^T B-fragment (32 keys x 16 queries, f16) ----
            v16h bp;
#pragma unroll
            for (int r = 0; r < 8; ++r) {
                const float x0 = __shfl_xor(st0[r], 16, 32);
                const float x1 = __shfl_xor(st1[r], 16, 32);
                if (hi == 0) {
                    bp[r]     = (_Float16)st0[r];   // key r
                    bp[8 + r] = (_Float16)x0;       // key 8+r
                } else {
                    bp[r]     = (_Float16)x1;       // key 16+r
                    bp[8 + r] = (_Float16)st1[r];   // key 24+r
                }
            }

            // ---- O^T += V^T . P^T over the 32-key chunk ----
#pragma unroll
            for (int dc = 0; dc < 4; ++dc)
                o[dc] = wmma_f16(load_a16(&Vlds[dc * 16 + ln][0], hi), bp, o[dc]);
        }
    }

    // ---- finalize: divide by softmax sum, store O ----
    const float inv = 1.0f / l_run;
    float* op = out + ((size_t)b * SEQ + qlane) * NSTATE + h * DHEAD;
#pragma unroll
    for (int dc = 0; dc < 4; ++dc) {
        float4 a = make_float4(o[dc][0] * inv, o[dc][1] * inv,
                               o[dc][2] * inv, o[dc][3] * inv);
        float4 c = make_float4(o[dc][4] * inv, o[dc][5] * inv,
                               o[dc][6] * inv, o[dc][7] * inv);
        float* p = op + dc * 16 + hi * 8;   // d = dc*16 + hi*8 + r
        *(float4*)(p)     = a;
        *(float4*)(p + 4) = c;
    }
}

extern "C" void kernel_launch(void* const* d_in, const int* in_sizes, int n_in,
                              void* d_out, int out_size, void* d_ws, size_t ws_size,
                              hipStream_t stream) {
    const float* x = (const float*)d_in[0];
    // d_in[1] (attn_mask) is pure causal -> implemented analytically, not read.
    float* out = (float*)d_out;
    const int B = in_sizes[0] / (SEQ * ROW3);
    dim3 grid(SEQ / 64, N_HEAD, B);
    flash_attn_causal_f16wmma<<<grid, dim3(128), 0, stream>>>(x, out);
}
